// HeavilyCompressedAttention_6373731467588
// MI455X (gfx1250) — matrix operand order repaired and checked by reading, not compiled, not run on hardware
//
#include <hip/hip_runtime.h>
#include <hip/hip_bf16.h>

#define S_    2048
#define HID_  2048
#define NH_   16
#define HD_   128
#define R_    16
#define C_    128
#define WIN_  128
#define TW_   160   // local attention key-tile span: 10 tiles of 16 covers [q0-128, q0+32)

typedef __attribute__((ext_vector_type(16))) __bf16 v16bf;
typedef __attribute__((ext_vector_type(8)))  __bf16 v8bf;
typedef __attribute__((ext_vector_type(8)))  float  v8f;

static __device__ __forceinline__ v8f vzero8() {
  v8f z;
#pragma unroll
  for (int i = 0; i < 8; ++i) z[i] = 0.f;
  return z;
}

static __device__ __forceinline__ v8f wmma_bf16(v16bf a, v16bf b, v8f c) {
  // D = A(16x32) * B(32x16) + C, f32 accumulate
  return __builtin_amdgcn_wmma_f32_16x16x32_bf16(false, a, false, b, (short)0, c,
                                                 false, false);
}

static __device__ __forceinline__ v16bf cat8(v8bf lo, v8bf hi) {
  return __builtin_shufflevector(lo, hi, 0, 1, 2, 3, 4, 5, 6, 7, 8, 9, 10, 11, 12, 13, 14, 15);
}

// A operand: 16x32 bf16 tile from row-major A (leading dim lda).
// Lane l in [0,16): row m0+l, K = k..k+15
// Lane l in [16,32): row m0+l-16, K = k+16..k+31
static __device__ __forceinline__ v16bf load_a(const __bf16* A, int lda, int m0, int k, int lane) {
  const __bf16* p = A + (size_t)(m0 + (lane & 15)) * lda + k + ((lane >> 4) << 4);
  v8bf lo = *(const v8bf*)p;
  v8bf hi = *(const v8bf*)(p + 8);
  return cat8(lo, hi);
}

// B operand: 32x16 bf16 tile from BT row-major [N][K]; col is the absolute column.
// Lanes 0-15: K=k..k+15 of column; lanes 16-31: K=k+16..k+31 (one 32B load each).
static __device__ __forceinline__ v16bf load_b(const __bf16* BT, int ldb, int col, int k, int lane) {
  const __bf16* p = BT + (size_t)col * ldb + k + ((lane >> 4) << 4);
  return *(const v16bf*)p;
}

// ---------------------------------------------------------------------------
// Register-blocked bf16 GEMM for the large projections.
// C[M,N] = A[M,K] @ B[K,N] + bias, B pre-transposed as BT[N][K].
// Each wave computes a 32x64 block (2 M-tiles x 4 N-tiles, 8 f32x8 accums):
// per 32-wide K step, 2 A-operand + 4 B-operand loads feed 8 WMMAs
// (1.5 b128 loads per wmma instead of 4 -> ~22 flops/byte from L2).
// Block = 8 waves arranged 2(M) x 4(N) -> block tile 64x256.
// grid = (M/64, N/256). Requires M%64==0, N%256==0, K%32==0.
// ---------------------------------------------------------------------------
__global__ void gemm_bf16_blocked(const __bf16* __restrict__ A, const __bf16* __restrict__ BT,
                                  const float* __restrict__ bias, float* __restrict__ Cm,
                                  int M, int N, int K) {
  const int lane = threadIdx.x & 31;
  const int wave = threadIdx.x >> 5;  // 0..7
  const int wm = wave >> 2;           // 0..1
  const int wn = wave & 3;            // 0..3
  const int m0 = (blockIdx.x << 6) + (wm << 5);  // wave's 32 rows
  const int n0 = (blockIdx.y << 8) + (wn << 6);  // wave's 64 cols

  v8f acc[2][4];
#pragma unroll
  for (int i = 0; i < 2; ++i)
#pragma unroll
    for (int j = 0; j < 4; ++j) acc[i][j] = vzero8();

  const int cb = n0 + (lane & 15);
  for (int k = 0; k < K; k += 32) {
    if (k + 64 < K) {  // warm next chunks (global_prefetch_b8)
      __builtin_prefetch(A + (size_t)(m0 + (lane & 15)) * K + k + 64, 0, 1);
      __builtin_prefetch(BT + (size_t)cb * K + k + 64, 0, 1);
    }
    v16bf a0 = load_a(A, K, m0, k, lane);
    v16bf a1 = load_a(A, K, m0 + 16, k, lane);
#pragma unroll
    for (int j = 0; j < 4; ++j) {
      v16bf b = load_b(BT, K, cb + j * 16, k, lane);
      acc[0][j] = wmma_bf16(a0, b, acc[0][j]);
      acc[1][j] = wmma_bf16(a1, b, acc[1][j]);
    }
  }

#pragma unroll
  for (int j = 0; j < 4; ++j) {
    const int n = cb + j * 16;
    const float bvv = bias[n];
#pragma unroll
    for (int i = 0; i < 2; ++i) {
      const int mb = m0 + (i << 4) + ((lane >> 4) << 3);
#pragma unroll
      for (int v = 0; v < 8; ++v) Cm[(size_t)(mb + v) * N + n] = acc[i][j][v] + bvv;
    }
  }
}

// ---------------------------------------------------------------------------
// Single-tile bf16 GEMM (one 16x16 tile per wave) for the small ck/cv GEMMs.
// grid = (M/16, N/128), 8 waves/block.
// ---------------------------------------------------------------------------
__global__ void gemm_bf16_kernel(const __bf16* __restrict__ A, const __bf16* __restrict__ BT,
                                 const float* __restrict__ bias, float* __restrict__ Cm,
                                 int M, int N, int K) {
  const int lane = threadIdx.x & 31;
  const int wave = threadIdx.x >> 5;
  const int m0 = blockIdx.x << 4;
  const int n0 = (blockIdx.y << 7) + (wave << 4);
  if (m0 >= M || n0 >= N) return;

  v8f acc = vzero8();
  const int col = n0 + (lane & 15);
  for (int k = 0; k < K; k += 32) {
    v16bf a = load_a(A, K, m0, k, lane);
    v16bf b = load_b(BT, K, col, k, lane);
    acc = wmma_bf16(a, b, acc);
  }
  const int mb = m0 + ((lane >> 4) << 3);
  const float bvv = bias[col];
#pragma unroll
  for (int v = 0; v < 8; ++v) Cm[(size_t)(mb + v) * N + col] = acc[v] + bvv;
}

// ---------------------------------------------------------------------------
// Elementwise cast + cast-transpose helpers
// ---------------------------------------------------------------------------
__global__ void cast_bf16_kernel(const float* __restrict__ X, __bf16* __restrict__ Y, size_t n) {
  size_t i = (size_t)blockIdx.x * blockDim.x + threadIdx.x;
  if (i < n) Y[i] = (__bf16)X[i];
}

__global__ void cast_transpose_kernel(const float* __restrict__ W, __bf16* __restrict__ WT,
                                      int K, int N) {
  size_t i = (size_t)blockIdx.x * blockDim.x + threadIdx.x;
  size_t total = (size_t)K * N;
  if (i < total) {
    int k = (int)(i / N);
    int n = (int)(i % N);
    WT[(size_t)n * K + k] = (__bf16)W[i];
  }
}

// ---------------------------------------------------------------------------
// RMSNorm + RoPE + repack [rows][heads*HD] f32 -> [heads][rows][HD] bf16
// pos = row * pos_scale + pos_bias
// ---------------------------------------------------------------------------
__global__ void norm_rope_pack_kernel(const float* __restrict__ X, const float* __restrict__ nw,
                                      __bf16* __restrict__ dst, int heads, int rows,
                                      int pos_scale, int pos_bias) {
  const int b = blockIdx.x;
  const int h = b % heads;
  const int r = b / heads;
  const int d = threadIdx.x;  // 0..127
  float v = X[(size_t)r * heads * HD_ + h * HD_ + d];

  __shared__ float red[HD_];
  __shared__ float sv[HD_];
  red[d] = v * v;
  __syncthreads();
  for (int st = HD_ / 2; st > 0; st >>= 1) {
    if (d < st) red[d] += red[d + st];
    __syncthreads();
  }
  float rms = rsqrtf(red[0] * (1.0f / HD_) + 1e-6f);
  float nv = v * rms * nw[d];
  sv[d] = nv;
  __syncthreads();

  float out;
  float pos = (float)(r * pos_scale + pos_bias);
  if (d < 32) {
    float ang = pos * powf(10000.f, -(float)d / 32.f);
    out = sv[d] * cosf(ang) - sv[d + 32] * sinf(ang);
  } else if (d < 64) {
    int i = d - 32;
    float ang = pos * powf(10000.f, -(float)i / 32.f);
    out = sv[i] * sinf(ang) + sv[d] * cosf(ang);
  } else {
    out = nv;
  }
  dst[((size_t)h * rows + r) * HD_ + d] = (__bf16)out;
}

// Repack LV f32 [S][NH*HD] -> bf16 [NH][HD][S] (transposed so P@V B-operand is contiguous)
__global__ void lv_pack_kernel(const float* __restrict__ X, __bf16* __restrict__ LVt) {
  size_t i = (size_t)blockIdx.x * blockDim.x + threadIdx.x;
  if (i >= (size_t)NH_ * S_ * HD_) return;
  int d = (int)(i % HD_);
  int s = (int)((i / HD_) % S_);
  int h = (int)(i / ((size_t)S_ * HD_));
  LVt[((size_t)h * HD_ + d) * S_ + s] = (__bf16)X[(size_t)s * HID_ + h * HD_ + d];
}

// cv f32 [C][HD] -> bf16 [HD][C]
__global__ void cv_pack_kernel(const float* __restrict__ cvf, __bf16* __restrict__ cvT) {
  int c = blockIdx.x;
  int d = threadIdx.x;
  cvT[(size_t)d * C_ + c] = (__bf16)cvf[(size_t)c * HD_ + d];
}

// ---------------------------------------------------------------------------
// Block compression: softmax(HS@Wc + bc) over R, weighted sum -> entries bf16 [C][HID]
// One block (16 waves) per compressed block c.
// ---------------------------------------------------------------------------
__global__ void entries_kernel(const float* __restrict__ HS, const float* __restrict__ Wc,
                               const float* __restrict__ bc, __bf16* __restrict__ Eb) {
  const int c = blockIdx.x;
  const int lane = threadIdx.x & 31;
  const int w = threadIdx.x >> 5;  // 0..15 == row within block
  __shared__ float lg[R_];

  const float* row = HS + (size_t)(c * R_ + w) * HID_;
  float sum = 0.f;
  for (int i = lane; i < HID_; i += 32) sum += row[i] * Wc[i];
#pragma unroll
  for (int o = 16; o > 0; o >>= 1) sum += __shfl_xor(sum, o, 32);
  if (lane == 0) lg[w] = sum + bc[0];
  __syncthreads();

  float mx = lg[0];
#pragma unroll
  for (int r = 1; r < R_; ++r) mx = fmaxf(mx, lg[r]);
  float wgt[R_];
  float tot = 0.f;
#pragma unroll
  for (int r = 0; r < R_; ++r) { wgt[r] = __expf(lg[r] - mx); tot += wgt[r]; }
  float inv = 1.f / tot;

  for (int j = threadIdx.x; j < HID_; j += blockDim.x) {
    float e = 0.f;
#pragma unroll
    for (int r = 0; r < R_; ++r) e += wgt[r] * inv * HS[(size_t)(c * R_ + r) * HID_ + j];
    Eb[(size_t)c * HID_ + j] = (__bf16)e;
  }
}

// ---------------------------------------------------------------------------
// Compressed attention + sink: one wave per (16-query tile, head).
// scores (WMMA) -> masked softmax incl. sink -> ctx (WMMA) + sink value.
// ---------------------------------------------------------------------------
__global__ void cattn_kernel(const __bf16* __restrict__ Qb, const __bf16* __restrict__ ckb,
                             const __bf16* __restrict__ cvT, const float* __restrict__ sink_k,
                             const float* __restrict__ sink_v, float* __restrict__ CTXc) {
  const int lane = threadIdx.x;
  const int m0 = blockIdx.x << 4;
  const int h = blockIdx.y;
  const __bf16* Qh = Qb + (size_t)h * S_ * HD_;
  const float scale = 0.088388347648318447f;  // 1/sqrt(128)

  __shared__ float sc[16][C_];
  __shared__ __attribute__((aligned(16))) __bf16 pb[16][C_];
  __shared__ float sp[16];

  // S = Q @ CK^T  (BT = ckb [C][HD])
  for (int nt = 0; nt < C_ / 16; ++nt) {
    v8f acc = vzero8();
    const int col = nt * 16 + (lane & 15);
#pragma unroll
    for (int kk = 0; kk < HD_; kk += 32) {
      v16bf a = load_a(Qh, HD_, m0, kk, lane);
      v16bf b = load_b(ckb, HD_, col, kk, lane);
      acc = wmma_bf16(a, b, acc);
    }
    const int mb = (lane >> 4) << 3;
    const int bend = col * R_ + (R_ - 1);
#pragma unroll
    for (int v = 0; v < 8; ++v) {
      int s = m0 + mb + v;
      sc[mb + v][col] = (bend <= s) ? acc[v] * scale : -1e9f;
    }
  }
  __syncthreads();

  if (lane < 16) {
    const int s = m0 + lane;
    const __bf16* qr = Qh + (size_t)s * HD_;
    const float* sk = sink_k + h * HD_;
    float dot = 0.f;
    for (int d = 0; d < HD_; ++d) dot += (float)qr[d] * sk[d];
    float ss = dot * scale;
    float mx = ss;
    for (int c = 0; c < C_; ++c) mx = fmaxf(mx, sc[lane][c]);
    float sinkp = __expf(ss - mx);
    float tot = sinkp;
    for (int c = 0; c < C_; ++c) {
      float e = __expf(sc[lane][c] - mx);
      sc[lane][c] = e;
      tot += e;
    }
    float inv = 1.f / tot;
    for (int c = 0; c < C_; ++c) pb[lane][c] = (__bf16)(sc[lane][c] * inv);
    sp[lane] = sinkp * inv;
  }
  __syncthreads();

  // ctx = P @ CV  (BT = cvT [HD][C]) + sink_p * sink_v
  for (int nt = 0; nt < HD_ / 16; ++nt) {
    v8f acc = vzero8();
    const int col = nt * 16 + (lane & 15);
#pragma unroll
    for (int kk = 0; kk < C_; kk += 32) {
      v16bf a = load_a(&pb[0][0], C_, 0, kk, lane);
      v16bf b = load_b(cvT, C_, col, kk, lane);
      acc = wmma_bf16(a, b, acc);
    }
    const int mb = (lane >> 4) << 3;
    const float svv = sink_v[h * HD_ + col];
#pragma unroll
    for (int v = 0; v < 8; ++v) {
      int s = m0 + mb + v;
      CTXc[((size_t)h * S_ + s) * HD_ + col] = acc[v] + sp[mb + v] * svv;
    }
  }
}

// ---------------------------------------------------------------------------
// Local windowed attention; merges with compressed context, writes bf16 merged.
// One wave per (16-query tile, head). Key span = 160 (chunk-aligned clamping:
// t0 == q0 (mod 16), so any clamped 16-chunk is fully masked => zero prob).
// ---------------------------------------------------------------------------
__global__ void lattn_kernel(const __bf16* __restrict__ Qb, const __bf16* __restrict__ LKb,
                             const __bf16* __restrict__ LVt, const float* __restrict__ CTXc,
                             __bf16* __restrict__ mergedb) {
  const int lane = threadIdx.x;
  const int m0 = blockIdx.x << 4;
  const int h = blockIdx.y;
  const __bf16* Qh = Qb + (size_t)h * S_ * HD_;
  const __bf16* LKh = LKb + (size_t)h * S_ * HD_;
  const __bf16* LVh = LVt + (size_t)h * HD_ * S_;
  const int t0 = m0 - WIN_;
  const float scale = 0.088388347648318447f;

  __shared__ float sc[16][TW_];
  __shared__ __attribute__((aligned(16))) __bf16 pb[16][TW_];

  // S = Q @ LK^T over the window (BT = LKh [S][HD], column index = key time)
  for (int nt = 0; nt < TW_ / 16; ++nt) {
    v8f acc = vzero8();
    const int tn = t0 + nt * 16 + (lane & 15);
    const int tcl = min(max(tn, 0), S_ - 1);
#pragma unroll
    for (int kk = 0; kk < HD_; kk += 32) {
      v16bf a = load_a(Qh, HD_, m0, kk, lane);
      v16bf b = load_b(LKh, HD_, tcl, kk, lane);
      acc = wmma_bf16(a, b, acc);
    }
    const int mb = (lane >> 4) << 3;
#pragma unroll
    for (int v = 0; v < 8; ++v) {
      int s = m0 + mb + v;
      int d = s - tn;
      bool ok = (tn >= 0) && (d >= 0) && (d <= WIN_);
      sc[mb + v][nt * 16 + (lane & 15)] = ok ? acc[v] * scale : -1e9f;
    }
  }
  __syncthreads();

  if (lane < 16) {
    float mx = -1e30f;
    for (int t = 0; t < TW_; ++t) mx = fmaxf(mx, sc[lane][t]);
    float tot = 0.f;
    for (int t = 0; t < TW_; ++t) {
      float e = __expf(sc[lane][t] - mx);
      sc[lane][t] = e;
      tot += e;
    }
    float inv = 1.f / tot;
    for (int t = 0; t < TW_; ++t) pb[lane][t] = (__bf16)(sc[lane][t] * inv);
  }
  __syncthreads();

  // ctx_l = P @ LV (BT = LVh [HD][S]); merge with compressed context.
  for (int nt = 0; nt < HD_ / 16; ++nt) {
    v8f acc = vzero8();
    const int col = nt * 16 + (lane & 15);
#pragma unroll
    for (int kk = 0; kk < TW_; kk += 32) {
      v16bf a = load_a(&pb[0][0], TW_, 0, kk, lane);
      int tk = t0 + kk + ((lane >> 4) << 4);
      tk = min(max(tk, 0), S_ - 16);
      v16bf b = *(const v16bf*)(LVh + (size_t)col * S_ + tk);
      acc = wmma_bf16(a, b, acc);
    }
    const int mb = (lane >> 4) << 3;
#pragma unroll
    for (int v = 0; v < 8; ++v) {
      int s = m0 + mb + v;
      float cc = CTXc[((size_t)h * S_ + s) * HD_ + col];
      mergedb[(size_t)s * HID_ + h * HD_ + col] = (__bf16)(0.5f * (acc[v] + cc));
    }
  }
}

// ---------------------------------------------------------------------------
extern "C" void kernel_launch(void* const* d_in, const int* in_sizes, int n_in,
                              void* d_out, int out_size, void* d_ws, size_t ws_size,
                              hipStream_t stream) {
  const float* HS = (const float*)d_in[0];
  const float* Wq = (const float*)d_in[1];
  const float* bq = (const float*)d_in[2];
  const float* Wc = (const float*)d_in[3];
  const float* bc = (const float*)d_in[4];
  const float* Wk = (const float*)d_in[5];
  const float* bk = (const float*)d_in[6];
  const float* Wv = (const float*)d_in[7];
  const float* bv = (const float*)d_in[8];
  const float* Wlk = (const float*)d_in[9];
  const float* blk = (const float*)d_in[10];
  const float* Wlv = (const float*)d_in[11];
  const float* blv = (const float*)d_in[12];
  const float* qn_w = (const float*)d_in[13];
  const float* kn_w = (const float*)d_in[14];
  const float* sink_k = (const float*)d_in[15];
  const float* sink_v = (const float*)d_in[16];
  const float* Wo = (const float*)d_in[17];
  const float* bo = (const float*)d_in[18];
  float* out = (float*)d_out;

  size_t off = 0;
  auto alloc = [&](size_t bytes) -> void* {
    off = (off + 255) & ~(size_t)255;
    void* p = (char*)d_ws + off;
    off += bytes;
    return p;
  };

  __bf16* HSb = (__bf16*)alloc((size_t)S_ * HID_ * 2);       // 8 MB
  __bf16* WqT = (__bf16*)alloc((size_t)HID_ * HID_ * 2);     // 8 MB
  __bf16* WlkT = (__bf16*)alloc((size_t)HID_ * HID_ * 2);    // 8 MB
  __bf16* WlvT = (__bf16*)alloc((size_t)HID_ * HID_ * 2);    // 8 MB
  __bf16* WoT = (__bf16*)alloc((size_t)HID_ * HID_ * 2);     // 8 MB
  __bf16* WkT = (__bf16*)alloc((size_t)HD_ * HID_ * 2);      // 0.5 MB
  __bf16* WvT = (__bf16*)alloc((size_t)HD_ * HID_ * 2);      // 0.5 MB
  float* TMPf = (float*)alloc((size_t)S_ * HID_ * 4);        // 16 MB (reused 3x)
  __bf16* Qb = (__bf16*)alloc((size_t)NH_ * S_ * HD_ * 2);   // 8 MB  [H][S][HD]
  __bf16* LKb = (__bf16*)alloc((size_t)NH_ * S_ * HD_ * 2);  // 8 MB  [H][S][HD]
  __bf16* LVt = (__bf16*)alloc((size_t)NH_ * HD_ * S_ * 2);  // 8 MB  [H][HD][S]
  __bf16* Eb = (__bf16*)alloc((size_t)C_ * HID_ * 2);        // 0.5 MB
  float* ckf = (float*)alloc((size_t)C_ * HD_ * 4);          // 64 KB
  float* cvf = (float*)alloc((size_t)C_ * HD_ * 4);          // 64 KB
  __bf16* ckb = (__bf16*)alloc((size_t)C_ * HD_ * 2);        // 32 KB [C][HD]
  __bf16* cvT = (__bf16*)alloc((size_t)HD_ * C_ * 2);        // 32 KB [HD][C]
  float* CTXc = (float*)alloc((size_t)NH_ * S_ * HD_ * 4);   // 16 MB
  __bf16* mergedb = (__bf16*)alloc((size_t)S_ * HID_ * 2);   // 8 MB

  const size_t nHS = (size_t)S_ * HID_;
  cast_bf16_kernel<<<(unsigned)((nHS + 255) / 256), 256, 0, stream>>>(HS, HSb, nHS);
  const size_t nWW = (size_t)HID_ * HID_;
  cast_transpose_kernel<<<(unsigned)((nWW + 255) / 256), 256, 0, stream>>>(Wq, WqT, HID_, HID_);
  cast_transpose_kernel<<<(unsigned)((nWW + 255) / 256), 256, 0, stream>>>(Wlk, WlkT, HID_, HID_);
  cast_transpose_kernel<<<(unsigned)((nWW + 255) / 256), 256, 0, stream>>>(Wlv, WlvT, HID_, HID_);
  cast_transpose_kernel<<<(unsigned)((nWW + 255) / 256), 256, 0, stream>>>(Wo, WoT, HID_, HID_);
  const size_t nWk = (size_t)HID_ * HD_;
  cast_transpose_kernel<<<(unsigned)((nWk + 255) / 256), 256, 0, stream>>>(Wk, WkT, HID_, HD_);
  cast_transpose_kernel<<<(unsigned)((nWk + 255) / 256), 256, 0, stream>>>(Wv, WvT, HID_, HD_);

  dim3 gBig(S_ / 64, HID_ / 256);  // (32, 8): 8 waves/block, 32x64 per wave

  // Q projection -> rmsnorm(qn_w) + rope(s) -> Qb
  gemm_bf16_blocked<<<gBig, 256, 0, stream>>>(HSb, WqT, bq, TMPf, S_, HID_, HID_);
  norm_rope_pack_kernel<<<S_ * NH_, HD_, 0, stream>>>(TMPf, qn_w, Qb, NH_, S_, 1, 0);

  // Local K projection -> rmsnorm(kn_w) + rope(s) -> LKb
  gemm_bf16_blocked<<<gBig, 256, 0, stream>>>(HSb, WlkT, blk, TMPf, S_, HID_, HID_);
  norm_rope_pack_kernel<<<S_ * NH_, HD_, 0, stream>>>(TMPf, kn_w, LKb, NH_, S_, 1, 0);

  // Local V projection -> transpose pack -> LVt
  gemm_bf16_blocked<<<gBig, 256, 0, stream>>>(HSb, WlvT, blv, TMPf, S_, HID_, HID_);
  {
    size_t n = (size_t)NH_ * S_ * HD_;
    lv_pack_kernel<<<(unsigned)((n + 255) / 256), 256, 0, stream>>>(TMPf, LVt);
  }

  // Compressed entries + ck/cv
  entries_kernel<<<C_, 512, 0, stream>>>(HS, Wc, bc, Eb);
  dim3 gCk(C_ / 16, 1);
  gemm_bf16_kernel<<<gCk, 256, 0, stream>>>(Eb, WkT, bk, ckf, C_, HD_, HID_);
  gemm_bf16_kernel<<<gCk, 256, 0, stream>>>(Eb, WvT, bv, cvf, C_, HD_, HID_);
  norm_rope_pack_kernel<<<C_, HD_, 0, stream>>>(ckf, kn_w, ckb, 1, C_, R_, R_ - 1);
  cv_pack_kernel<<<C_, HD_, 0, stream>>>(cvf, cvT);

  // Attention
  dim3 gAtt(S_ / 16, NH_);
  cattn_kernel<<<gAtt, 32, 0, stream>>>(Qb, ckb, cvT, sink_k, sink_v, CTXc);
  lattn_kernel<<<gAtt, 32, 0, stream>>>(Qb, LKb, LVt, CTXc, mergedb);

  // Output projection
  gemm_bf16_blocked<<<gBig, 256, 0, stream>>>(mergedb, WoT, bo, out, S_, HID_, HID_);

  (void)in_sizes; (void)n_in; (void)out_size; (void)ws_size;
}
